// NAM_31499290149001
// MI455X (gfx1250) — matrix operand order repaired and checked by reading, not compiled, not run on hardware
//
#include <hip/hip_runtime.h>

#define F_   256
#define H0_  128
#define H1_  128
#define H2_  64
#define OUT_ 16
#define B_   8192
#define BM_  128

typedef __attribute__((ext_vector_type(16))) __bf16 v16bf;
typedef __attribute__((ext_vector_type(8)))  float  v8f;

union Frag { v16bf v; uint4 q[2]; };

// native bf16 convert (RNE) -> v_cvt_pk_bf16_f32
__device__ __forceinline__ unsigned short f2bf(float f) {
  union { __bf16 b; unsigned short u; } x;
  x.b = (__bf16)f;
  return x.u;
}
__device__ __forceinline__ float bf2f(unsigned short u) {
  union { unsigned u; float f; } x; x.u = ((unsigned)u) << 16;
  return x.f;
}

// ---------------- weight-norm precompute: out = bf16( g * v / ||v||_rowwise ) ---
__global__ void nam_wnorm(const float* __restrict__ v, const float* __restrict__ g,
                          unsigned short* __restrict__ out, int nrows, int rowlen) {
  int row = blockIdx.x * blockDim.x + threadIdx.x;
  if (row >= nrows) return;
  const float* vr = v + (size_t)row * rowlen;
  float ss = 0.f;
  for (int i = 0; i < rowlen; ++i) { float t = vr[i]; ss += t * t; }
  float s = g[row] * rsqrtf(ss);
  unsigned short* o = out + (size_t)row * rowlen;
  for (int i = 0; i < rowlen; ++i) o[i] = f2bf(vr[i] * s);
}

// ---------------- Tensor Data Mover staging (CDNA5 TDM) ----------------------
#if defined(__has_builtin)
#if __has_builtin(__builtin_amdgcn_tensor_load_to_lds)
#define NAM_HAVE_TDM 1
#endif
#endif

#ifdef NAM_HAVE_TDM
typedef unsigned int u32x4v __attribute__((ext_vector_type(4)));
typedef int          i32x4v __attribute__((ext_vector_type(4)));
typedef int          i32x8v __attribute__((ext_vector_type(8)));

// 2-D bf16 tile load, global -> LDS, with row padding applied by the TDM.
//  width: elements per row, rows: #rows, padIntervalCode: pad every 2^(c+1) DWORDs,
//  padAmountCode: (c+1) DWORDs of pad.  D# layout per CDNA5 ISA §8.3/§8.4.
__device__ __forceinline__ void tdm_load_2d(const void* gsrc, unsigned ldsOff,
                                            int width, int rows,
                                            int padIntervalCode, int padAmountCode) {
  unsigned long long ga = (unsigned long long)gsrc;
  u32x4v g0;
  g0[0] = 1u;                                           // count=1, user descriptor
  g0[1] = ldsOff;                                       // lds_addr (bytes)
  g0[2] = (unsigned)(ga & 0xFFFFFFFFu);                 // global_addr[31:0]
  g0[3] = (unsigned)((ga >> 32) & 0x1FFFFFFu) | (2u << 30);  // [56:32] | type=2
  i32x8v g1;
  g1[0] = (1 << 16)                                     // data_size = 2 bytes
        | (1 << 20)                                     // pad_enable
        | (padIntervalCode << 22)
        | (padAmountCode << 25);
  g1[1] = (width & 0xFFFF) << 16;                       // tensor_dim0[15:0]
  g1[2] = ((unsigned)width >> 16) | ((rows & 0xFFFF) << 16);   // dim0 hi | dim1 lo
  g1[3] = ((unsigned)rows >> 16) | ((width & 0xFFFF) << 16);   // dim1 hi | tile_dim0
  g1[4] = (rows & 0xFFFF);                              // tile_dim1 (tile_dim2=0)
  g1[5] = width;                                        // tensor_dim0_stride lo32
  g1[6] = 0;
  g1[7] = 0;
  i32x4v z4 = {0, 0, 0, 0};
#if __clang_major__ >= 23
  i32x8v z8 = {0, 0, 0, 0, 0, 0, 0, 0};
  __builtin_amdgcn_tensor_load_to_lds(g0, g1, z4, z4, z8, 0);
#else
  __builtin_amdgcn_tensor_load_to_lds(g0, g1, z4, z4, 0);
#endif
}
#endif  // NAM_HAVE_TDM

// ---------------- WMMA helpers (A/B fragments per CDNA5 16-bit layouts) ------
// A-fragments for a wave's 16-row stripe are N-tile invariant: load them once
// into registers (KT Frags) so the nt-loop only streams B fragments from LDS.
template<int KT>
__device__ __forceinline__ void load_afrags(const unsigned short* aRow, int hsel,
                                            Frag* a) {
  #pragma unroll
  for (int kt = 0; kt < KT; ++kt) {
    const int ka = kt * 32 + hsel * 8;    // A: K chunks [hsel*8,+8) and [16+hsel*8,+8)
    a[kt].q[0] = *(const uint4*)(aRow + ka);
    a[kt].q[1] = *(const uint4*)(aRow + ka + 16);
  }
}

template<int KT>
__device__ __forceinline__ v8f wmma_bloop(const Frag* a, const unsigned short* bRow,
                                          int hsel, v8f acc) {
  #pragma unroll
  for (int kt = 0; kt < KT; ++kt) {
    Frag b;
    const int kb = kt * 32 + hsel * 8;    // B: K chunks [hsel*8,+8) and [16+hsel*8,+8)
    b.q[0] = *(const uint4*)(bRow + kb);
    b.q[1] = *(const uint4*)(bRow + kb + 16);
    acc = __builtin_amdgcn_wmma_f32_16x16x32_bf16(false, a[kt].v, false, b.v,
                                                  (short)0, acc, false, false);
  }
  return acc;
}

// ---------------- main fused per-feature MLP kernel --------------------------
// grid: (B/BM, F); block: 256 threads (8 wave32).  Dynamic LDS layout (bytes):
//  sW1 @0      128x136 bf16 = 34816
//  sW2 @34816   64x136 bf16 = 17408
//  sWo @52224   16x72  bf16 =  2304
//  sH0 @54528  128x136 bf16 = 34816
//  sH1 @89344  128x136 bf16 = 34816
//  sH2 @124160 128x72  bf16 = 18432
//  sX  @142592 128 f32      =   512   -> total 143104 B
__global__ __launch_bounds__(256, 1) void nam_main(
    const float* __restrict__ tab,
    const unsigned short* __restrict__ w0n, const float* __restrict__ b_in,
    const unsigned short* __restrict__ W1n, const float* __restrict__ b_h0,
    const unsigned short* __restrict__ W2n, const float* __restrict__ b_h1,
    const unsigned short* __restrict__ Won,
    float* __restrict__ dout) {
  extern __shared__ char smem[];
  unsigned short* sW1 = (unsigned short*)(smem + 0);
  unsigned short* sW2 = (unsigned short*)(smem + 34816);
  unsigned short* sWo = (unsigned short*)(smem + 52224);
  unsigned short* sH0 = (unsigned short*)(smem + 54528);
  unsigned short* sH1 = (unsigned short*)(smem + 89344);
  unsigned short* sH2 = (unsigned short*)(smem + 124160);
  float*          sX  = (float*)(smem + 142592);

  const int t     = threadIdx.x;
  const int f     = blockIdx.y;
  const int mbase = blockIdx.x * BM_;
  const int wv = t >> 5, lane = t & 31;

  // ---- stage this feature's weights into LDS ----
#ifdef NAM_HAVE_TDM
  // One TDM descriptor per weight matrix, issued by waves 0/1/2.  Row padding
  // (+16B) is applied by the TDM: W1/W2 rows are 64 DWORDs -> pad 4 DWORDs
  // every 64 (codes 5,3); Wout rows are 32 DWORDs -> codes 4,3.
  if (wv == 0) {
    tdm_load_2d(W1n + (size_t)f * H1_ * H0_, 0u,     H0_, H1_,  5, 3);
    __builtin_amdgcn_s_wait_tensorcnt(0);
  } else if (wv == 1) {
    tdm_load_2d(W2n + (size_t)f * H2_ * H1_, 34816u, H1_, H2_,  5, 3);
    __builtin_amdgcn_s_wait_tensorcnt(0);
  } else if (wv == 2) {
    tdm_load_2d(Won + (size_t)f * OUT_ * H2_, 52224u, H2_, OUT_, 4, 3);
    __builtin_amdgcn_s_wait_tensorcnt(0);
  }
#else
  {
    int row = t >> 1, hh = t & 1;                     // 128 rows x 2 halves
    const uint4* src = (const uint4*)(W1n + ((size_t)(f * H1_ + row) * H0_ + hh * 64));
    uint4* dst = (uint4*)(sW1 + row * 136 + hh * 64);
    #pragma unroll
    for (int i = 0; i < 8; ++i) dst[i] = src[i];
  }
  if (t < 64) {
    const uint4* src = (const uint4*)(W2n + (size_t)(f * H2_ + t) * H1_);
    uint4* dst = (uint4*)(sW2 + t * 136);
    #pragma unroll
    for (int i = 0; i < 16; ++i) dst[i] = src[i];
  }
  if (t < 16) {
    const uint4* src = (const uint4*)(Won + (size_t)(f * OUT_ + t) * H2_);
    uint4* dst = (uint4*)(sWo + t * 72);
    #pragma unroll
    for (int i = 0; i < 8; ++i) dst[i] = src[i];
  }
#endif
  if (t < BM_) {
    int b = mbase + t;
    float val = tab[((size_t)b * 2 + 0) * F_ + f];
    float mis = tab[((size_t)b * 2 + 1) * F_ + f];
    sX[t] = val * (1.0f - mis);
  }
  __syncthreads();

  // ---- input layer: h0[m,i] = relu(x[m]*w0[i] + b_in[i])  (outer product, VALU) ----
  {
    int m = t >> 1, hh = t & 1;
    float x = sX[m];
    const unsigned short* wp = w0n + (size_t)f * H0_;
    const float*          bp = b_in + (size_t)f * H0_;
    int i0 = hh * 64;
    #pragma unroll 8
    for (int i = i0; i < i0 + 64; ++i) {
      float h = fmaxf(fmaf(x, bf2f(wp[i]), bp[i]), 0.f);
      sH0[m * 136 + i] = f2bf(h);
    }
  }
  __syncthreads();

  const int r = lane & 15, hsel = lane >> 4;

  // ---- GEMM1: sH1 = relu(sH0(128x128) @ W1^T + b_h0), K=128 ----
  {
    Frag a[4];
    load_afrags<4>(sH0 + (wv * 16 + r) * 136, hsel, a);
    #pragma unroll
    for (int nt = 0; nt < 8; ++nt) {
      v8f acc = {};
      acc = wmma_bloop<4>(a, sW1 + (nt * 16 + r) * 136, hsel, acc);
      float bias = b_h0[(size_t)f * H1_ + nt * 16 + r];
      #pragma unroll
      for (int e = 0; e < 8; ++e) {
        int m = wv * 16 + hsel * 8 + e;
        sH1[m * 136 + nt * 16 + r] = f2bf(fmaxf(acc[e] + bias, 0.f));
      }
    }
  }
  __syncthreads();

  // ---- GEMM2: sH2 = relu(sH1(128x128) @ W2^T + b_h1), N=64, K=128 ----
  {
    Frag a[4];
    load_afrags<4>(sH1 + (wv * 16 + r) * 136, hsel, a);
    #pragma unroll
    for (int nt = 0; nt < 4; ++nt) {
      v8f acc = {};
      acc = wmma_bloop<4>(a, sW2 + (nt * 16 + r) * 136, hsel, acc);
      float bias = b_h1[(size_t)f * H2_ + nt * 16 + r];
      #pragma unroll
      for (int e = 0; e < 8; ++e) {
        int m = wv * 16 + hsel * 8 + e;
        sH2[m * 72 + nt * 16 + r] = f2bf(fmaxf(acc[e] + bias, 0.f));
      }
    }
  }
  __syncthreads();

  // ---- GEMM3: outputs(128x16) = sH2(128x64) @ Wo^T, K=64, store to global ----
  {
    Frag a[2];
    load_afrags<2>(sH2 + (wv * 16 + r) * 72, hsel, a);
    v8f acc = {};
    acc = wmma_bloop<2>(a, sWo + r * 72, hsel, acc);
    float* outs = dout + (size_t)B_ * OUT_;   // outputs region after logits
    #pragma unroll
    for (int e = 0; e < 8; ++e) {
      int m = wv * 16 + hsel * 8 + e;
      size_t b = (size_t)(mbase + m);
      outs[(b * F_ + f) * OUT_ + r] = acc[e];
    }
  }
}

// ---------------- logits[b,o] = sum_f outputs[b,f,o] + bias[o] ----------------
__global__ void nam_logits(const float* __restrict__ outs, const float* __restrict__ bias,
                           float* __restrict__ logits) {
  int idx = blockIdx.x * blockDim.x + threadIdx.x;   // exactly B*OUT threads
  int o = idx & (OUT_ - 1);
  int b = idx >> 4;
  float s = bias[o];
  const float* p = outs + (size_t)b * F_ * OUT_ + o;
  #pragma unroll 8
  for (int fe = 0; fe < F_; ++fe) s += p[(size_t)fe * OUT_];
  logits[idx] = s;
}

extern "C" void kernel_launch(void* const* d_in, const int* in_sizes, int n_in,
                              void* d_out, int out_size, void* d_ws, size_t ws_size,
                              hipStream_t stream) {
  const float* tab   = (const float*)d_in[0];
  const float* v_in  = (const float*)d_in[1];
  const float* g_in  = (const float*)d_in[2];
  const float* b_in  = (const float*)d_in[3];
  const float* v_h0  = (const float*)d_in[4];
  const float* g_h0  = (const float*)d_in[5];
  const float* b_h0  = (const float*)d_in[6];
  const float* v_h1  = (const float*)d_in[7];
  const float* g_h1  = (const float*)d_in[8];
  const float* b_h1  = (const float*)d_in[9];
  const float* v_out = (const float*)d_in[10];
  const float* g_out = (const float*)d_in[11];
  const float* bias  = (const float*)d_in[12];

  // workspace: bf16 weight-normed weights
  unsigned short* w0n = (unsigned short*)d_ws;                 // F*H0
  unsigned short* W1n = w0n + (size_t)F_ * H0_;                // F*H1*H0
  unsigned short* W2n = W1n + (size_t)F_ * H1_ * H0_;          // F*H2*H1
  unsigned short* Won = W2n + (size_t)F_ * H2_ * H1_;          // F*OUT*H2

  nam_wnorm<<<(F_ * H0_  + 255) / 256, 256, 0, stream>>>(v_in,  g_in,  w0n, F_ * H0_,  1);
  nam_wnorm<<<(F_ * H1_  + 255) / 256, 256, 0, stream>>>(v_h0,  g_h0,  W1n, F_ * H1_,  H0_);
  nam_wnorm<<<(F_ * H2_  + 255) / 256, 256, 0, stream>>>(v_h1,  g_h1,  W2n, F_ * H2_,  H1_);
  nam_wnorm<<<(F_ * OUT_ + 255) / 256, 256, 0, stream>>>(v_out, g_out, Won, F_ * OUT_, H2_);

  dim3 grid(B_ / BM_, F_);
  nam_main<<<grid, 256, 143104, stream>>>(tab, w0n, b_in, W1n, b_h0, W2n, b_h1, Won,
                                          (float*)d_out);

  nam_logits<<<(B_ * OUT_) / 256, 256, 0, stream>>>((const float*)d_out + (size_t)B_ * OUT_,
                                                    bias, (float*)d_out);
}
